// Fast2Order_DE_Conv_56427280335472
// MI455X (gfx1250) — matrix operand order repaired and checked by reading, not compiled, not run on hardware
//
#include <hip/hip_runtime.h>

// ---------------------------------------------------------------------------
// Fast 2nd-order conv == batched GEMM: out[b,o,l] = sum_f W[o,f]*feat[b,l,f]
// feat[b,l, c*81 + i*9 + j] = p[l,c,i]*p[l,c,j], p from 3x3 patches of x.
// M=128, K=1296 (pad->1312), N=3844 (62x62), B=16.  bf16 WMMA, f32 accum.
// Per block: 8x8 output tile, 8 waves x 16 O-rows, feat chunk ping-pong in LDS.
// ---------------------------------------------------------------------------

typedef __attribute__((ext_vector_type(16))) __bf16        v16bf;
typedef __attribute__((ext_vector_type(8)))  float         v8f;
typedef __attribute__((ext_vector_type(4)))  unsigned int  v4u;

#define IN_CH  16
#define OUT_CH 128
#define HDIM   64
#define WDIM   64
#define HO     62
#define WO     62
#define FTOT   1296           // IN_CH * 81
#define KCHUNK 32
#define NKC    41             // ceil(1296/32) -> K padded to 1312
#define FROW   48             // padded feat row stride (halfs), 96B = 16B aligned

union Frag {
    v16bf v;
    v4u   q[2];
};

__device__ __forceinline__ unsigned short f2bf(float f) {
    unsigned int u = __float_as_uint(f);
    unsigned int r = u + 0x7FFFu + ((u >> 16) & 1u);   // round-to-nearest-even
    return (unsigned short)(r >> 16);
}

// Pre-pass: W (f32, [128][1296]) -> Wb (bf16) swizzled as [NKC][128][32],
// zero padded for f >= 1296, so A-fragments are contiguous 16B loads.
__global__ __launch_bounds__(256) void pack_w_bf16(const float* __restrict__ W,
                                                   unsigned short* __restrict__ Wb) {
    int e  = blockIdx.x * 256 + threadIdx.x;   // < NKC*128*32 = 167936
    int kc = e >> 12;                          // / 4096
    int r  = e & 4095;
    int m  = r >> 5;
    int kk = r & 31;
    int f  = kc * KCHUNK + kk;
    float v = (f < FTOT) ? W[(size_t)m * FTOT + f] : 0.0f;
    Wb[e] = f2bf(v);
}

__global__ __launch_bounds__(256) void f2o_conv_wmma(const float* __restrict__ x,
                                                     const unsigned short* __restrict__ Wb,
                                                     float* __restrict__ out) {
    __shared__ float xsf[IN_CH * 100];                             // 16ch x 10x10 window
    __shared__ __align__(16) unsigned short featL[2][64][FROW];    // ping-pong bf16 feat

    const int tx   = blockIdx.x;            // 8 tiles in x
    const int ty   = blockIdx.y;            // 8 tiles in y
    const int b    = blockIdx.z;            // 16 batches
    const int x0   = tx * 8, y0 = ty * 8;
    const int tid  = threadIdx.x;
    const int lane = tid & 31;
    const int wave = tid >> 5;              // 8 waves, each owns 16 O-rows
    const int hi   = lane >> 4;
    const int lq   = lane & 15;

    // Stage the 16ch x 10x10 input window (clamped; edge tiles produce values
    // only for locations we never store).
    for (int e = tid; e < IN_CH * 100; e += 256) {
        int c  = e / 100;
        int r  = e - c * 100;
        int yy = r / 10, xx = r - yy * 10;
        int gy = y0 + yy; if (gy > HDIM - 1) gy = HDIM - 1;
        int gx = x0 + xx; if (gx > WDIM - 1) gx = WDIM - 1;
        xsf[e] = x[((size_t)(b * IN_CH + c) * HDIM + gy) * WDIM + gx];
    }

    v8f acc[4] = {v8f{}, v8f{}, v8f{}, v8f{}};

    // Feat staging: each thread owns one k of the chunk and 8 consecutive n's,
    // which all share one output row ly -> two 8-float consecutive xs runs.
    const int kk      = tid & 31;
    const int nb      = (tid >> 5) * 8;
    const int rowbase = (nb >> 3) * 10;     // ly * 10 (lx = 0..7)

    // Incrementally tracked decomposition of f = kc*32 + kk:
    //   cc = f / 81, rr = f % 81   (f advances by 32 < 81 per chunk)
    int cc = 0, rr = kk;

    // Stage chunk `buf` from current (cc, rr) state.
    auto stage = [&](int buf) {
        if (cc < IN_CH) {
            int i  = rr / 9;
            int j  = rr - 9 * i;
            int i3 = i / 3, j3 = j / 3;
            int d1 = i3 * 10 + (i - 3 * i3);            // ri*10 + rj
            int d2 = j3 * 10 + (j - 3 * j3);            // si*10 + sj
            const float* p1 = &xsf[cc * 100 + rowbase + d1];
            const float* p2 = &xsf[cc * 100 + rowbase + d2];
#pragma unroll
            for (int u = 0; u < 8; ++u)
                featL[buf][nb + u][kk] = f2bf(p1[u] * p2[u]);
        } else {
#pragma unroll
            for (int u = 0; u < 8; ++u) featL[buf][nb + u][kk] = 0;
        }
        // advance to next chunk's (cc, rr)
        rr += KCHUNK;
        if (rr >= 81) { rr -= 81; cc += 1; }
    };

    __syncthreads();           // xs window visible
    stage(0);                  // chunk 0 -> buffer 0

    for (int kc = 0; kc < NKC; ++kc) {
        __syncthreads();       // chunk kc staged; prior reads of buf[(kc+1)&1] done

        if (kc + 1 < NKC) stage((kc + 1) & 1);   // overlap next-chunk staging

        // A fragment (bf16 16x32): row m = wave*16 + lq;
        // elems 0..7 = K hi*8+0..7, elems 8..15 = K 16+hi*8+0..7.
        Frag a;
        const unsigned short* wrow =
            Wb + ((size_t)kc * OUT_CH + wave * 16 + lq) * KCHUNK;
        a.q[0] = *(const v4u*)(wrow + hi * 8);
        a.q[1] = *(const v4u*)(wrow + 16 + hi * 8);

        const unsigned short(*fb)[FROW] = featL[kc & 1];
#pragma unroll
        for (int nt = 0; nt < 4; ++nt) {
            // B fragment (bf16 32x16): column n = nt*16+lq; elems 0..7 = K hi*8+0..7, elems 8..15 = K 16+hi*8+0..7.
            Frag bf;
            const unsigned short* frow = &fb[nt * 16 + lq][hi * 8];
            bf.q[0] = *(const v4u*)(frow);
            bf.q[1] = *(const v4u*)(frow + 16);

            acc[nt] = __builtin_amdgcn_wmma_f32_16x16x32_bf16(
                false, a.v, false, bf.v, (short)0, acc[nt], false, false);
        }
    }

    // C/D layout: VGPR e -> M = e + 8*hi, N = lq.  Stream stores (no reuse).
#pragma unroll
    for (int nt = 0; nt < 4; ++nt) {
        int n  = nt * 16 + lq;
        int yy = y0 + (n >> 3);
        int xx = x0 + (n & 7);
        if (yy < HO && xx < WO) {
#pragma unroll
            for (int e = 0; e < 8; ++e) {
                int o = wave * 16 + hi * 8 + e;
                __builtin_nontemporal_store(
                    acc[nt][e],
                    &out[((size_t)(b * OUT_CH + o) * HO + yy) * WO + xx]);
            }
        }
    }
}

extern "C" void kernel_launch(void* const* d_in, const int* in_sizes, int n_in,
                              void* d_out, int out_size, void* d_ws, size_t ws_size,
                              hipStream_t stream) {
    (void)in_sizes; (void)n_in; (void)out_size; (void)ws_size;

    const float* x = (const float*)d_in[0];   // (16,16,64,64) f32
    const float* W = (const float*)d_in[1];   // (128,1296) f32
    float* out = (float*)d_out;               // (16,128,62,62) f32
    unsigned short* Wb = (unsigned short*)d_ws;  // 41*128*32 bf16 = 328 KB scratch

    pack_w_bf16<<<(NKC * OUT_CH * KCHUNK) / 256, 256, 0, stream>>>(W, Wb);

    dim3 grid(8, 8, 16);
    f2o_conv_wmma<<<grid, 256, 0, stream>>>(x, Wb, out);
}
